// GNN_45707041964838
// MI455X (gfx1250) — matrix operand order repaired and checked by reading, not compiled, not run on hardware
//
#include <hip/hip_runtime.h>
#include <hip/hip_bf16.h>

typedef _Float16 half8 __attribute__((ext_vector_type(8)));
typedef _Float16 v16h  __attribute__((ext_vector_type(16)));
typedef float    v8f   __attribute__((ext_vector_type(8)));

#define DFEAT 128
#define LDS_STRIDE 136   // halves per row; 272B keeps 16B alignment for b128 LDS ops

// ---------------- utility kernels ----------------

__global__ void GNN_zero_f32(float* __restrict__ p, int n) {
    int i = blockIdx.x * blockDim.x + threadIdx.x;
    int stride = gridDim.x * blockDim.x;
    for (; i < n; i += stride) p[i] = 0.0f;
}

__global__ void GNN_count_edges(const int* __restrict__ dst, float* __restrict__ cnt, int nEdges) {
    int e = blockIdx.x * blockDim.x + threadIdx.x;
    if (e < nEdges) atomicAdd(&cnt[dst[e]], 1.0f);
}

__global__ void GNN_make_inv(float* __restrict__ cnt, int n) {
    int i = blockIdx.x * blockDim.x + threadIdx.x;
    if (i < n) cnt[i] = 1.0f / fmaxf(cnt[i], 1.0f);
}

__global__ void GNN_conv_w_f16(const float* __restrict__ w, _Float16* __restrict__ o, int n) {
    int i = blockIdx.x * blockDim.x + threadIdx.x;
    if (i < n) o[i] = (_Float16)w[i];
}

// ---------------- scatter: wave per edge, float4 per lane ----------------

__global__ void GNN_scatter(const float* __restrict__ feats, const int* __restrict__ src,
                            const int* __restrict__ dst, float* __restrict__ agg, int nEdges) {
    const int e    = blockIdx.x * 8 + (threadIdx.x >> 5);
    const int lane = threadIdx.x & 31;
    if (e >= nEdges) return;
    const int s = src[e];
    const int d = dst[e];
    const float4 v = *(const float4*)(feats + (size_t)s * DFEAT + lane * 4);
    float* o = agg + (size_t)d * DFEAT + lane * 4;
    atomicAdd(o + 0, v.x);
    atomicAdd(o + 1, v.y);
    atomicAdd(o + 2, v.z);
    atomicAdd(o + 3, v.w);
}

// ---------------- fused SAGE GEMM: out = (agg*inv)@Wl^T + x@Wr^T + b ----------------
// 256 threads = 8 waves; block covers 16 node rows x 128 output cols.
// Wave w computes the 16x16 tile at column base w*16 via v_wmma_f32_16x16x32_f16.

__global__ __launch_bounds__(256)
void GNN_sage_gemm(const float* __restrict__ agg, const float* __restrict__ inv,
                   const float* __restrict__ xin,
                   const _Float16* __restrict__ Wl, const _Float16* __restrict__ Wr,
                   const float* __restrict__ bias, float* __restrict__ out,
                   int nNodes, int doLeaky) {
    __shared__ _Float16 sM[16][LDS_STRIDE];  // mean rows, f16
    __shared__ _Float16 sX[16][LDS_STRIDE];  // x rows, f16

    const int t = threadIdx.x;
    const int nodeBase = blockIdx.x * 16;

    // ---- load 16 rows of (agg*inv) and x, convert to f16, stage in LDS ----
    {
        const int row = t >> 4;          // 0..15
        const int cb  = (t & 15) * 8;    // 0,8,...,120
        const int node = nodeBase + row;
        float4 a0 = {0,0,0,0}, a1 = {0,0,0,0}, x0 = {0,0,0,0}, x1 = {0,0,0,0};
        float s = 0.0f;
        if (node < nNodes) {
            s = inv[node];
            const float4* ap = (const float4*)(agg + (size_t)node * DFEAT + cb);
            a0 = ap[0]; a1 = ap[1];
            const float4* xp = (const float4*)(xin + (size_t)node * DFEAT + cb);
            x0 = xp[0]; x1 = xp[1];
        }
        half8 hm, hx;
        hm[0] = (_Float16)(a0.x * s); hm[1] = (_Float16)(a0.y * s);
        hm[2] = (_Float16)(a0.z * s); hm[3] = (_Float16)(a0.w * s);
        hm[4] = (_Float16)(a1.x * s); hm[5] = (_Float16)(a1.y * s);
        hm[6] = (_Float16)(a1.z * s); hm[7] = (_Float16)(a1.w * s);
        hx[0] = (_Float16)x0.x; hx[1] = (_Float16)x0.y;
        hx[2] = (_Float16)x0.z; hx[3] = (_Float16)x0.w;
        hx[4] = (_Float16)x1.x; hx[5] = (_Float16)x1.y;
        hx[6] = (_Float16)x1.z; hx[7] = (_Float16)x1.w;
        *(half8*)&sM[row][cb] = hm;
        *(half8*)&sX[row][cb] = hx;
    }
    __syncthreads();

    const int wid  = t >> 5;        // wave id 0..7 -> column tile
    const int lane = t & 31;
    const int n16  = lane & 15;     // A-row / B-col within tile
    const int kh   = lane >> 4;     // K-half select per ISA layout
    const int col  = wid * 16 + n16;

    const _Float16* wl = Wl + (size_t)col * DFEAT;
    const _Float16* wr = Wr + (size_t)col * DFEAT;

    v8f acc = {};
#pragma unroll
    for (int k0 = 0; k0 < DFEAT; k0 += 32) {
        // A fragment (16x32 f16): lane holds A[M][k0+8*kh .. +7] and A[M][k0+16+8*kh .. +7]
        half8 alo = *(const half8*)&sM[n16][k0 + 8 * kh];
        half8 ahi = *(const half8*)&sM[n16][k0 + 16 + 8 * kh];
        v16h aM = __builtin_shufflevector(alo, ahi, 0,1,2,3,4,5,6,7,8,9,10,11,12,13,14,15);
        // B fragment (32x16 f16): lane holds W[col][k0+8*kh .. +7] and W[col][k0+16+8*kh .. +7] (B = W^T, row-major W over K)
        half8 blo = *(const half8*)(wl + k0 + 8 * kh);
        half8 bhi = *(const half8*)(wl + k0 + 16 + 8 * kh);
        v16h bL = __builtin_shufflevector(blo, bhi, 0,1,2,3,4,5,6,7,8,9,10,11,12,13,14,15);
        acc = __builtin_amdgcn_wmma_f32_16x16x32_f16(false, aM, false, bL, (short)0, acc, false, false);

        half8 xlo = *(const half8*)&sX[n16][k0 + 8 * kh];
        half8 xhi = *(const half8*)&sX[n16][k0 + 16 + 8 * kh];
        v16h aX = __builtin_shufflevector(xlo, xhi, 0,1,2,3,4,5,6,7,8,9,10,11,12,13,14,15);
        half8 rlo = *(const half8*)(wr + k0 + 8 * kh);
        half8 rhi = *(const half8*)(wr + k0 + 16 + 8 * kh);
        v16h bR = __builtin_shufflevector(rlo, rhi, 0,1,2,3,4,5,6,7,8,9,10,11,12,13,14,15);
        acc = __builtin_amdgcn_wmma_f32_16x16x32_f16(false, aX, false, bR, (short)0, acc, false, false);
    }

    // ---- epilogue: bias + optional leaky relu; C/D layout: VGPR j -> row j + 8*kh, col = lane
    const float bv = bias[col];
#pragma unroll
    for (int j = 0; j < 8; ++j) {
        const int node = nodeBase + j + 8 * kh;
        if (node < nNodes) {
            float v = acc[j] + bv;
            if (doLeaky) v = (v > 0.0f) ? v : 0.2f * v;
            out[(size_t)node * DFEAT + col] = v;
        }
    }
}

// ---------------- host-side launch ----------------

extern "C" void kernel_launch(void* const* d_in, const int* in_sizes, int n_in,
                              void* d_out, int out_size, void* d_ws, size_t ws_size,
                              hipStream_t stream) {
    (void)n_in; (void)out_size; (void)ws_size;

    const float* x   = (const float*)d_in[0];
    const int*   ei  = (const int*)d_in[1];
    const float* W1l = (const float*)d_in[2];
    const float* b1  = (const float*)d_in[3];
    const float* W1r = (const float*)d_in[4];
    const float* W2l = (const float*)d_in[5];
    const float* b2  = (const float*)d_in[6];
    const float* W2r = (const float*)d_in[7];
    float* out = (float*)d_out;

    const int N = in_sizes[0] / DFEAT;
    const int E = in_sizes[1] / 2;
    const int* src = ei;       // edge_index[0, :]
    const int* dst = ei + E;   // edge_index[1, :]

    // workspace carve (256B aligned): agg | h | inv | 4x f16 weights  (~52 MB)
    char* ws = (char*)d_ws;
    size_t off = 0;
    auto carve = [&](size_t bytes) {
        void* p = ws + off;
        off = (off + bytes + 255) & ~(size_t)255;
        return p;
    };
    float*     agg = (float*)carve((size_t)N * DFEAT * sizeof(float));
    float*     h   = (float*)carve((size_t)N * DFEAT * sizeof(float));
    float*     inv = (float*)carve((size_t)N * sizeof(float));
    _Float16*  w1l = (_Float16*)carve((size_t)DFEAT * DFEAT * sizeof(_Float16));
    _Float16*  w1r = (_Float16*)carve((size_t)DFEAT * DFEAT * sizeof(_Float16));
    _Float16*  w2l = (_Float16*)carve((size_t)DFEAT * DFEAT * sizeof(_Float16));
    _Float16*  w2r = (_Float16*)carve((size_t)DFEAT * DFEAT * sizeof(_Float16));

    const int nFeatTotal = N * DFEAT;
    const int gridScatter = (E + 7) / 8;        // wave (32 lanes) per edge
    const int gridGemm    = (N + 15) / 16;      // 16-node tiles
    const int nW = DFEAT * DFEAT;

    // degree counts -> inverse (shared across both layers)
    GNN_zero_f32<<<256, 256, 0, stream>>>(inv, N);
    GNN_zero_f32<<<2048, 256, 0, stream>>>(agg, nFeatTotal);
    GNN_count_edges<<<(E + 255) / 256, 256, 0, stream>>>(dst, inv, E);
    GNN_make_inv<<<(N + 255) / 256, 256, 0, stream>>>(inv, N);

    // f16 weight staging (L2-resident, 128KB total)
    GNN_conv_w_f16<<<(nW + 255) / 256, 256, 0, stream>>>(W1l, w1l, nW);
    GNN_conv_w_f16<<<(nW + 255) / 256, 256, 0, stream>>>(W1r, w1r, nW);
    GNN_conv_w_f16<<<(nW + 255) / 256, 256, 0, stream>>>(W2l, w2l, nW);
    GNN_conv_w_f16<<<(nW + 255) / 256, 256, 0, stream>>>(W2r, w2r, nW);

    // layer 1: scatter-sum, then fused mean/GEMM/bias/leaky-relu -> h
    GNN_scatter<<<gridScatter, 256, 0, stream>>>(x, src, dst, agg, E);
    GNN_sage_gemm<<<gridGemm, 256, 0, stream>>>(agg, inv, x, w1l, w1r, b1, h, N, 1);

    // layer 2: re-zero agg, scatter h, fused GEMM -> out (no activation)
    GNN_zero_f32<<<2048, 256, 0, stream>>>(agg, nFeatTotal);
    GNN_scatter<<<gridScatter, 256, 0, stream>>>(h, src, dst, agg, E);
    GNN_sage_gemm<<<gridGemm, 256, 0, stream>>>(agg, inv, h, w2l, w2r, b2, out, N, 0);
}
